// FactorizedAttention_54408645706394
// MI455X (gfx1250) — matrix operand order repaired and checked by reading, not compiled, not run on hardware
//
#include <hip/hip_runtime.h>
#include <math.h>

typedef __attribute__((ext_vector_type(16))) _Float16 v16h;
typedef __attribute__((ext_vector_type(8)))  _Float16 v8h;
typedef __attribute__((ext_vector_type(8)))  float    v8f;

union AFrag { v16h v; v8h h[2]; };

// ---------------- constants of this problem ----------------
#define CB   8        // batch
#define CC   384      // channels
#define CN   6400     // tokens (80*80)
#define CNK  400      // downsampled tokens (20*20)
#define CNH  8        // heads
#define CHD  48       // head dim

// ---------------- elementwise f32 -> f16 ----------------
__global__ void fa_cvt_f16(const float* __restrict__ src, _Float16* __restrict__ dst, long n) {
    long i = (long)blockIdx.x * blockDim.x + threadIdx.x;
    if (i < n) dst[i] = (_Float16)src[i];
}

// ---------------- depthwise 4x4 stride-4 conv ----------------
// x: [B, N, C] (NLC), kvw: [C,1,4,4] ; out kv: [B, NK, C] (f32 + f16)
__global__ void fa_dwconv(const float* __restrict__ x, const float* __restrict__ kvw,
                          float* __restrict__ kv32, _Float16* __restrict__ kv16) {
    long idx = (long)blockIdx.x * blockDim.x + threadIdx.x;
    const long total = (long)CB * CNK * CC;
    if (idx >= total) return;
    int c   = (int)(idx % CC);
    long t  = idx / CC;
    int nk  = (int)(t % CNK);
    int b   = (int)(t / CNK);
    int hk  = nk / 20, wk = nk % 20;
    const float* xb = x + ((long)b * CN) * CC + c;
    const float* wc = kvw + c * 16;
    float s = 0.f;
#pragma unroll
    for (int i = 0; i < 4; ++i)
#pragma unroll
        for (int j = 0; j < 4; ++j)
            s += xb[(long)((hk * 4 + i) * 80 + (wk * 4 + j)) * CC] * wc[i * 4 + j];
    kv32[idx] = s;
    kv16[idx] = (_Float16)s;
}

// ---------------- generic WMMA GEMM: out = (A @ W^T + bias) * scale ----------------
// A: f16 [M x 384] row-major, W: f16 [384 x 384] row-major ([n_out, k]).
// One wave -> 32x64 output block (2 M-tiles x 4 N-tiles): every B fragment
// feeds two WMMAs. N = K = 384 fixed. M must be a multiple of 32.
__global__ void fa_wmma_gemm_n384(const _Float16* __restrict__ A,
                                  const _Float16* __restrict__ W,
                                  const float* __restrict__ bias,
                                  float* __restrict__ out32,
                                  _Float16* __restrict__ out16,
                                  int M, float scale) {
    int wid  = (int)((blockIdx.x * blockDim.x + threadIdx.x) >> 5);
    int lane = threadIdx.x & 31;
    int mb = wid / 6;           // 32-row block index
    int ns = wid - mb * 6;      // 64-col strip index
    if (mb * 32 >= M) return;   // wave-uniform: EXEC stays all-ones for WMMA
    int r  = lane & 15;
    int hf = lane >> 4;

    const _Float16* arow0 = A + (long)(mb * 32 + r) * CC;
    const _Float16* arow1 = arow0 + (long)16 * CC;
    v8f acc[2][4] = {};
#pragma unroll 2
    for (int kk = 0; kk < CC; kk += 32) {
        AFrag a0, a1;
        a0.h[0] = *(const v8h*)(arow0 + kk + 8 * hf);        // K = kk + 8*hf + 0..7
        a0.h[1] = *(const v8h*)(arow0 + kk + 16 + 8 * hf);   // K = kk + 16 + 8*hf + 0..7
        a1.h[0] = *(const v8h*)(arow1 + kk + 8 * hf);
        a1.h[1] = *(const v8h*)(arow1 + kk + 16 + 8 * hf);
#pragma unroll
        for (int t = 0; t < 4; ++t) {
            const _Float16* wrow = W + (long)(ns * 64 + 16 * t + r) * CC + kk + 8 * hf;
            AFrag bfr;
            bfr.h[0] = *(const v8h*)(wrow);                  // K = kk + 8*hf + 0..7
            bfr.h[1] = *(const v8h*)(wrow + 16);             // K = kk + 16 + 8*hf + 0..7
            acc[0][t] = __builtin_amdgcn_wmma_f32_16x16x32_f16(
                false, a0.v, false, bfr.v, (short)0, acc[0][t], false, false);
            acc[1][t] = __builtin_amdgcn_wmma_f32_16x16x32_f16(
                false, a1.v, false, bfr.v, (short)0, acc[1][t], false, false);
        }
    }
#pragma unroll
    for (int t = 0; t < 4; ++t) {
        int col = ns * 64 + 16 * t + r;
        float bv = bias ? bias[col] : 0.0f;
#pragma unroll
        for (int mi = 0; mi < 2; ++mi)
#pragma unroll
            for (int vv = 0; vv < 8; ++vv) {
                int rowo = mb * 32 + mi * 16 + vv + 8 * hf;
                float val = (acc[mi][t][vv] + bv) * scale;
                long o = (long)rowo * CC + col;
                if (out32) out32[o] = val;
                if (out16) out16[o] = (_Float16)val;
            }
    }
}

// ---------------- softmax stats over the 400 key tokens, per (b, channel) ----------------
__global__ void fa_softmax_stats(const float* __restrict__ k32,
                                 float* __restrict__ maxv, float* __restrict__ rsum) {
    int idx = blockIdx.x * blockDim.x + threadIdx.x;     // [0, B*C)
    if (idx >= CB * CC) return;
    int b = idx / CC, c = idx - b * CC;
    const float* col = k32 + (long)b * CNK * CC + c;
    float m = -1e30f;
    for (int n = 0; n < CNK; ++n) m = fmaxf(m, col[n * CC]);
    float s = 0.f;
    for (int n = 0; n < CNK; ++n) s += expf(col[n * CC] - m);
    maxv[idx] = m;
    rsum[idx] = 1.0f / s;
}

// in-place: k32 <- exp(k - max) / sum  (the softmax'd keys)
__global__ void fa_softmax_apply(float* __restrict__ k32,
                                 const float* __restrict__ maxv,
                                 const float* __restrict__ rsum) {
    long idx = (long)blockIdx.x * blockDim.x + threadIdx.x;
    const long total = (long)CB * CNK * CC;
    if (idx >= total) return;
    int c = (int)(idx % CC);
    int b = (int)(idx / ((long)CNK * CC));
    int sc = b * CC + c;
    k32[idx] = expf(k32[idx] - maxv[sc]) * rsum[sc];
}

// ---------------- kvmat[b,h,i,j] = sum_n ksm[b,n,h*48+i] * v[b,n,h*48+j] ----------------
// stored transposed+padded as W-style weights: kvmatT16[(b*8+h)][j][i], i padded to 64 with 0
__global__ void fa_kvmat(const float* __restrict__ ksm, const float* __restrict__ v32,
                         _Float16* __restrict__ kvmatT16) {
    int idx = blockIdx.x * blockDim.x + threadIdx.x;     // [0, 64*48*64)
    const int total = CB * CNH * CHD * 64;
    if (idx >= total) return;
    int i  = idx & 63;
    int t  = idx >> 6;
    int j  = t % CHD;
    int bh = t / CHD;
    if (i >= CHD) { kvmatT16[idx] = (_Float16)0.0f; return; }  // zero K-padding
    int b = bh >> 3, h = bh & 7;
    int ci = h * CHD + i, cj = h * CHD + j;
    const float* kb = ksm + (long)b * CNK * CC;
    const float* vb = v32 + (long)b * CNK * CC;
    float acc = 0.f;
    for (int n = 0; n < CNK; ++n)
        acc += kb[(long)n * CC + ci] * vb[(long)n * CC + cj];
    kvmatT16[idx] = (_Float16)acc;
}

// ---------------- attn = q_scaled @ kvmat per (b,h) : [6400x48] @ [48x48], K padded to 64 ----
// One wave -> 32x48 output block (2 M-tiles x 3 N-tiles).
__global__ void fa_wmma_attn(const _Float16* __restrict__ q16,
                             const _Float16* __restrict__ kvmatT16,
                             _Float16* __restrict__ attn16) {
    int wid  = (int)((blockIdx.x * blockDim.x + threadIdx.x) >> 5);
    int lane = threadIdx.x & 31;
    int bh = wid / (CN / 32);            // 0..63
    int mb = wid - bh * (CN / 32);       // 0..199
    if (bh >= CB * CNH) return;          // wave-uniform
    int b = bh >> 3, h = bh & 7;
    int r  = lane & 15;
    int hf = lane >> 4;

    const _Float16* arow0 = q16 + ((long)(b * CN + mb * 32 + r)) * CC + h * CHD;
    const _Float16* arow1 = arow0 + (long)16 * CC;
    const _Float16* Wb    = kvmatT16 + (long)bh * CHD * 64;
    v8f acc[2][3] = {};
#pragma unroll
    for (int kk = 0; kk < 64; kk += 32) {   // K padded 48 -> 64 (zero weights)
        AFrag a0, a1;
        a0.h[0] = *(const v8h*)(arow0 + kk + 8 * hf);
        a0.h[1] = *(const v8h*)(arow0 + kk + 16 + 8 * hf);
        a1.h[0] = *(const v8h*)(arow1 + kk + 8 * hf);
        a1.h[1] = *(const v8h*)(arow1 + kk + 16 + 8 * hf);
#pragma unroll
        for (int t = 0; t < 3; ++t) {
            const _Float16* wrow = Wb + (long)(16 * t + r) * 64 + kk + 8 * hf;
            AFrag bfr;
            bfr.h[0] = *(const v8h*)(wrow);
            bfr.h[1] = *(const v8h*)(wrow + 16);
            acc[0][t] = __builtin_amdgcn_wmma_f32_16x16x32_f16(
                false, a0.v, false, bfr.v, (short)0, acc[0][t], false, false);
            acc[1][t] = __builtin_amdgcn_wmma_f32_16x16x32_f16(
                false, a1.v, false, bfr.v, (short)0, acc[1][t], false, false);
        }
    }
#pragma unroll
    for (int t = 0; t < 3; ++t)
#pragma unroll
        for (int mi = 0; mi < 2; ++mi)
#pragma unroll
            for (int vv = 0; vv < 8; ++vv) {
                long rowo = (long)b * CN + mb * 32 + mi * 16 + vv + 8 * hf;
                attn16[rowo * CC + h * CHD + 16 * t + r] = (_Float16)acc[mi][t][vv];
            }
}

// =================================================================
extern "C" void kernel_launch(void* const* d_in, const int* in_sizes, int n_in,
                              void* d_out, int out_size, void* d_ws, size_t ws_size,
                              hipStream_t stream) {
    const float* x      = (const float*)d_in[0];
    const float* q_w    = (const float*)d_in[1];
    const float* q_b    = (const float*)d_in[2];
    const float* kv_w   = (const float*)d_in[3];
    const float* k_w    = (const float*)d_in[4];
    const float* v_w    = (const float*)d_in[5];
    const float* v_b    = (const float*)d_in[6];
    const float* proj_w = (const float*)d_in[7];
    const float* proj_b = (const float*)d_in[8];
    float* out = (float*)d_out;

    const long XE  = (long)CB * CN * CC;     // 19,660,800
    const long KVE = (long)CB * CNK * CC;    // 1,228,800
    const long WE  = (long)CC * CC;          // 147,456

    // carve workspace (aligned to 256B)
    char* w = (char*)d_ws;
    auto carve = [&](size_t bytes) -> void* {
        void* p = (void*)w;
        w += (bytes + 255) & ~(size_t)255;
        return p;
    };
    _Float16* x16   = (_Float16*)carve(XE * 2 + 256);   // reused as attn16 later
    _Float16* q16   = (_Float16*)carve(XE * 2 + 256);   // +pad: attn K-padding over-reads
    _Float16* qw16  = (_Float16*)carve(WE * 2);
    _Float16* kw16  = (_Float16*)carve(WE * 2);
    _Float16* vw16  = (_Float16*)carve(WE * 2);
    _Float16* pw16  = (_Float16*)carve(WE * 2);
    float*    kv32  = (float*)carve(KVE * 4);
    _Float16* kv16  = (_Float16*)carve(KVE * 2);
    float*    k32   = (float*)carve(KVE * 4);
    float*    v32   = (float*)carve(KVE * 4);
    float*    maxv  = (float*)carve((size_t)CB * CC * 4);
    float*    rsum  = (float*)carve((size_t)CB * CC * 4);
    _Float16* kvmT  = (_Float16*)carve((size_t)CB * CNH * CHD * 64 * 2);
    _Float16* attn16 = x16;   // alias: x16 no longer needed once q is built

    const float scale = 1.0f / sqrtf((float)CHD);

    // zero the over-read pad past q16 (attn kernel reads up to +16 f16 past last row)
    hipMemsetAsync((char*)q16 + XE * 2, 0, 256, stream);

    // 1) f32 -> f16 conversions
    fa_cvt_f16<<<dim3((unsigned)((XE + 255) / 256)), dim3(256), 0, stream>>>(x, x16, XE);
    fa_cvt_f16<<<dim3((unsigned)((WE + 255) / 256)), dim3(256), 0, stream>>>(q_w, qw16, WE);
    fa_cvt_f16<<<dim3((unsigned)((WE + 255) / 256)), dim3(256), 0, stream>>>(k_w, kw16, WE);
    fa_cvt_f16<<<dim3((unsigned)((WE + 255) / 256)), dim3(256), 0, stream>>>(v_w, vw16, WE);
    fa_cvt_f16<<<dim3((unsigned)((WE + 255) / 256)), dim3(256), 0, stream>>>(proj_w, pw16, WE);

    // 2) depthwise downsampling conv -> kv_l
    fa_dwconv<<<dim3((unsigned)((KVE + 255) / 256)), dim3(256), 0, stream>>>(x, kv_w, kv32, kv16);

    // 3) q = (x @ q_w^T + q_b) * scale   [scale folded in]  (51200x384, K=384)
    {
        int waves = (CB * CN / 32) * 6;   // 9600
        fa_wmma_gemm_n384<<<dim3(waves / 8), dim3(256), 0, stream>>>(
            x16, qw16, q_b, nullptr, q16, CB * CN, scale);
    }
    // 4) k = kv @ k_w^T (no bias, fp32 out for softmax); v = kv @ v_w^T + v_b
    {
        int waves = (CB * CNK / 32) * 6;  // 600
        fa_wmma_gemm_n384<<<dim3(waves / 8), dim3(256), 0, stream>>>(
            kv16, kw16, nullptr, k32, nullptr, CB * CNK, 1.0f);
        fa_wmma_gemm_n384<<<dim3(waves / 8), dim3(256), 0, stream>>>(
            kv16, vw16, v_b, v32, nullptr, CB * CNK, 1.0f);
    }
    // 5) softmax over key tokens per (b, channel)
    fa_softmax_stats<<<dim3((CB * CC + 127) / 128), dim3(128), 0, stream>>>(k32, maxv, rsum);
    fa_softmax_apply<<<dim3((unsigned)((KVE + 255) / 256)), dim3(256), 0, stream>>>(k32, maxv, rsum);

    // 6) kvmat per head, stored transposed + K-padded as f16 weights
    {
        int total = CB * CNH * CHD * 64;  // 196,608
        fa_kvmat<<<dim3((total + 255) / 256), dim3(256), 0, stream>>>(k32, v32, kvmT);
    }
    // 7) attn = q_scaled @ kvmat  (per (b,h), WMMA, K padded to 64)
    {
        int waves = CB * CNH * (CN / 32); // 12800
        fa_wmma_attn<<<dim3(waves / 8), dim3(256), 0, stream>>>(q16, kvmT, attn16);
    }
    // 8) out = attn @ proj_w^T + proj_b  (fp32 to d_out)
    {
        int waves = (CB * CN / 32) * 6;   // 9600
        fa_wmma_gemm_n384<<<dim3(waves / 8), dim3(256), 0, stream>>>(
            attn16, pw16, proj_b, out, nullptr, CB * CN, 1.0f);
    }
    (void)in_sizes; (void)n_in; (void)out_size; (void)ws_size;
}
